// RecursiverLayer_81810537054472
// MI455X (gfx1250) — matrix operand order repaired and checked by reading, not compiled, not run on hardware
//
#include <hip/hip_runtime.h>

typedef __attribute__((ext_vector_type(16))) _Float16 v16h;
typedef __attribute__((ext_vector_type(8)))  float    v8f;

static constexpr int   NROW  = 8192;
static constexpr int   FDIM  = 256;
static constexpr float ALPHA = 0.2f;
static constexpr float NEGV  = -9000000000000000.0f;

// Swizzled f16 weight layout: [mat(2)][tile48][kc(8)][lane(32)][t(16)]
//   element = W_mat[tile*16 + (lane&15)][kc*32 + (lane>>4)*16 + t]
static constexpr int PER_MAT   = 48 * 8 * 32 * 16;   // 196608 halfs per matrix
static constexpr int CBUF_H    = 6 * 8 * 32 * 16;    // 24576 halfs = 48 KB per c-iter

__device__ __forceinline__ v8f wmma16(v16h a, v16h b, v8f c) {
    return __builtin_amdgcn_wmma_f32_16x16x32_f16(
        false, a, false, b, (short)0, c, false, false);
}

__device__ __forceinline__ float sigmoidf_fast(float x) {
    return 1.0f / (1.0f + __expf(-x));
}

// ---------------------------------------------------------------------------
// Stage 0: one-shot convert + swizzle of W_ih / W_hh into WMMA B-fragment
// order (f32 -> f16). 1.5 MB read / 768 KB write — noise.
// ---------------------------------------------------------------------------
__global__ __launch_bounds__(256) void swizzle_weights(
    const float* __restrict__ W_ih, const float* __restrict__ W_hh,
    _Float16* __restrict__ Wsw)
{
    const int idx = blockIdx.x * 256 + threadIdx.x;      // 2*PER_MAT total
    const int mat = idx / PER_MAT;
    const int r   = idx % PER_MAT;
    const int t    = r & 15;
    const int lane = (r >> 4) & 31;
    const int kc   = (r >> 9) & 7;
    const int tile = r >> 12;                            // 0..47
    const int row = tile * 16 + (lane & 15);
    const int k   = kc * 32 + (lane >> 4) * 16 + t;
    const float* W = mat ? W_hh : W_ih;
    Wsw[idx] = (_Float16)W[(size_t)row * FDIM + k];
}

// ---------------------------------------------------------------------------
// Stage 1: fused gather + GRU (WMMA f16) + a1/a2 projection.
// 64 blocks x 8 waves; each wave owns a 16-row tile. B fragments are staged
// block-wide through LDS with async global->LDS copies, double-buffered so
// iteration c+1's weights stream in (ASYNCcnt) while c's WMMAs run off LDS.
// ---------------------------------------------------------------------------
__global__ __launch_bounds__(256) void gru_proj_stage1(
    const float* __restrict__ inputs, const _Float16* __restrict__ Wsw,
    const float* __restrict__ b_ih,   const float* __restrict__ b_hh,
    const float* __restrict__ avec,   const int* __restrict__ nbr,
    float* __restrict__ s1, float* __restrict__ s2)
{
    __shared__ _Float16 smem[2 * CBUF_H];                // 96 KB (320 KB WGP LDS)

    const int lane = threadIdx.x & 31;
    const int wave = threadIdx.x >> 5;
    const int tile = blockIdx.x * 8 + wave;      // 16-row tile (512 total)
    const int row0 = tile * 16;

    const int hk = lane >> 4;                    // lane half (0/1)
    const int ml = lane & 15;                    // A row / B+C column in tile

    // --- A fragments: gather x (nbr[:,1]) and h (nbr[:,0]) rows, f32->f16 ---
    const int arow = row0 + ml;
    const float* xrow = inputs + (size_t)nbr[2 * arow + 1] * FDIM;
    const float* hrow = inputs + (size_t)nbr[2 * arow + 0] * FDIM;

    v16h Ax[8], Ah[8];
    #pragma unroll
    for (int kc = 0; kc < 8; ++kc) {
        const int kb = kc * 32;
        #pragma unroll
        for (int t = 0; t < 8; ++t) {
            // k order used here (contiguous, same as Wsw): lanes 0-15 K={0..15}, lanes 16-31 K={16..31}
            Ax[kc][t]     = (_Float16)xrow[kb + hk * 16 + t];
            Ax[kc][t + 8] = (_Float16)xrow[kb + hk * 16 + 8 + t];
            Ah[kc][t]     = (_Float16)hrow[kb + hk * 16 + t];
            Ah[kc][t + 8] = (_Float16)hrow[kb + hk * 16 + 8 + t];
        }
    }

    // h source rows for the C-layout rows this lane owns (M = r + 8*hk)
    int hsrcC[8];
    #pragma unroll
    for (int r = 0; r < 8; ++r)
        hsrcC[r] = nbr[2 * (row0 + r + 8 * hk) + 0];

    float s1p[8], s2p[8];
    #pragma unroll
    for (int r = 0; r < 8; ++r) { s1p[r] = 0.f; s2p[r] = 0.f; }

    // Cooperative async copy of iteration c's 48 KB of B fragments into buf.
    // Per gate g: 8 KB contiguous region of Wsw; 512 B128 chunks / 256 threads.
    auto issue_copies = [&](int c, int buf) {
        _Float16* dst = smem + buf * CBUF_H;
        #pragma unroll
        for (int g = 0; g < 6; ++g) {
            const int mat = g / 3;
            const int t48 = (g % 3) * 16 + c;
            const _Float16* src = Wsw + (size_t)((mat * 48 + t48) * 8) * 512;
            const unsigned ldsbase =
                (unsigned)(uintptr_t)(dst + g * 4096);   // low 32 bits = LDS offset
            #pragma unroll
            for (int i = 0; i < 2; ++i) {
                const int chunk = threadIdx.x + i * 256;         // 0..511
                const unsigned loff = ldsbase + chunk * 16;
                const _Float16* gp  = src + chunk * 8;
                asm volatile("global_load_async_to_lds_b128 %0, %1, off"
                             :: "v"(loff), "v"(gp) : "memory");
            }
        }
    };

    // Prologue: fill buffer 0 for c=0.
    issue_copies(0, 0);
    asm volatile("s_wait_asynccnt 0x0" ::: "memory");
    __syncthreads();

    for (int c = 0; c < 16; ++c) {               // 16 gate-column tiles
        // Prefetch next iteration's weights into the other buffer.
        if (c + 1 < 16) issue_copies(c + 1, (c + 1) & 1);

        const _Float16* bufp = smem + (c & 1) * CBUF_H;
        auto fragL = [&](int g, int kc) -> v16h {
            return *(const v16h*)(bufp + ((g * 8 + kc) * 32 + lane) * 16);
        };

        v8f air = {}, aiz = {}, ain = {}, ahr = {}, ahz = {}, ahn = {};
        #pragma unroll
        for (int kc = 0; kc < 8; ++kc) {
            air = wmma16(Ax[kc], fragL(0, kc), air);
            aiz = wmma16(Ax[kc], fragL(1, kc), aiz);
            ain = wmma16(Ax[kc], fragL(2, kc), ain);
            ahr = wmma16(Ah[kc], fragL(3, kc), ahr);
            ahz = wmma16(Ah[kc], fragL(4, kc), ahz);
            ahn = wmma16(Ah[kc], fragL(5, kc), ahn);
        }

        // --- gates in C layout: lane owns column col, rows M = r + 8*hk ---
        const int col = c * 16 + ml;
        const float bir = b_ih[col], biz = b_ih[256 + col], bin = b_ih[512 + col];
        const float bhr = b_hh[col], bhz = b_hh[256 + col], bhn = b_hh[512 + col];
        const float a1c = avec[col], a2c = avec[FDIM + col];

        #pragma unroll
        for (int r = 0; r < 8; ++r) {
            const float rg = sigmoidf_fast(air[r] + ahr[r] + bir + bhr);
            const float zg = sigmoidf_fast(aiz[r] + ahz[r] + biz + bhz);
            const float ng = tanhf(ain[r] + bin + rg * (ahn[r] + bhn));
            const float hv = inputs[(size_t)hsrcC[r] * FDIM + col];
            const float o  = (1.0f - zg) * ng + zg * hv;
            s1p[r] += o * a1c;
            s2p[r] += o * a2c;
        }

        // Own async copies done, then block-wide visibility for next iter.
        asm volatile("s_wait_asynccnt 0x0" ::: "memory");
        __syncthreads();
    }

    // --- reduce across the 16 lanes sharing each row; write s1/s2 ---
    #pragma unroll
    for (int r = 0; r < 8; ++r) {
        float v1 = s1p[r], v2 = s2p[r];
        #pragma unroll
        for (int m = 1; m < 16; m <<= 1) {
            v1 += __shfl_xor(v1, m, 32);
            v2 += __shfl_xor(v2, m, 32);
        }
        if (ml == 0) {
            s1[row0 + 8 * hk + r] = v1;
            s2[row0 + 8 * hk + r] = v2;
        }
    }
}

// ---------------------------------------------------------------------------
// Stage 2: fused leaky_relu + mask + row softmax. One block per row.
// adj read once (256 MB), out written once (256 MB); s2 stays in L2.
// ---------------------------------------------------------------------------
static constexpr int S2T = 512;
static constexpr int CPT = NROW / S2T;   // 16 columns per thread

__global__ __launch_bounds__(S2T) void mask_softmax_stage2(
    const float* __restrict__ s1, const float* __restrict__ s2,
    const int* __restrict__ adj, float* __restrict__ out)
{
    const int row = blockIdx.x;
    const int tid = threadIdx.x;
    const float si = s1[row];
    const int* arow = adj + (size_t)row * NROW;
    float*     orow = out + (size_t)row * NROW;

    __shared__ float red[16];

    float e[CPT];
    float mx = NEGV;
    #pragma unroll
    for (int t = 0; t < CPT; ++t) {
        const int j = tid + t * S2T;              // coalesced stride
        const float v  = si + s2[j];
        float ev = (v > 0.0f) ? v : (ALPHA * v);
        ev = (arow[j] > 0) ? ev : NEGV;
        e[t] = ev;
        mx = fmaxf(mx, ev);
    }

    const int lane = tid & 31, w = tid >> 5;
    #pragma unroll
    for (int m = 16; m >= 1; m >>= 1) mx = fmaxf(mx, __shfl_xor(mx, m, 32));
    if (lane == 0) red[w] = mx;
    __syncthreads();
    if (tid < 16) {
        float v = red[tid];
        #pragma unroll
        for (int m = 8; m >= 1; m >>= 1) v = fmaxf(v, __shfl_xor(v, m, 32));
        if (tid == 0) red[0] = v;
    }
    __syncthreads();
    mx = red[0];
    __syncthreads();

    float sum = 0.0f;
    #pragma unroll
    for (int t = 0; t < CPT; ++t) {
        const float p = __expf(e[t] - mx);
        e[t] = p;
        sum += p;
    }
    #pragma unroll
    for (int m = 16; m >= 1; m >>= 1) sum += __shfl_xor(sum, m, 32);
    if (lane == 0) red[w] = sum;
    __syncthreads();
    if (tid < 16) {
        float v = red[tid];
        #pragma unroll
        for (int m = 8; m >= 1; m >>= 1) v += __shfl_xor(v, m, 32);
        if (tid == 0) red[0] = v;
    }
    __syncthreads();
    const float inv = 1.0f / red[0];

    #pragma unroll
    for (int t = 0; t < CPT; ++t) {
        const int j = tid + t * S2T;
        orow[j] = e[t] * inv;
    }
}

// ---------------------------------------------------------------------------
extern "C" void kernel_launch(void* const* d_in, const int* in_sizes, int n_in,
                              void* d_out, int out_size, void* d_ws, size_t ws_size,
                              hipStream_t stream) {
    const float* inputs = (const float*)d_in[0];
    const float* W_ih   = (const float*)d_in[1];
    const float* W_hh   = (const float*)d_in[2];
    const float* b_ih   = (const float*)d_in[3];
    const float* b_hh   = (const float*)d_in[4];
    const float* avec   = (const float*)d_in[5];
    const int*   nbr    = (const int*)d_in[6];
    const int*   adj    = (const int*)d_in[7];
    float* out = (float*)d_out;

    float*    s1  = (float*)d_ws;                       // [NROW]
    float*    s2  = s1 + NROW;                          // [NROW]
    _Float16* Wsw = (_Float16*)(s2 + NROW);             // 2*PER_MAT halfs (768 KB)

    swizzle_weights<<<(2 * PER_MAT) / 256, 256, 0, stream>>>(W_ih, W_hh, Wsw);
    gru_proj_stage1<<<NROW / (16 * 8), 256, 0, stream>>>(
        inputs, Wsw, b_ih, b_hh, avec, nbr, s1, s2);
    mask_softmax_stage2<<<NROW, S2T, 0, stream>>>(s1, s2, adj, out);
}
